// GIN_NP_31756988186809
// MI455X (gfx1250) — compile-verified, matrix operand order checked by reading, not run on hardware
//
#include <hip/hip_runtime.h>
#include <hip/hip_bf16.h>
#include <stdint.h>

typedef unsigned short u16;
typedef unsigned int   u32;
typedef __attribute__((ext_vector_type(16))) __bf16 v16bf;
typedef __attribute__((ext_vector_type(8)))  float  v8f;

#define HID 256
#define VOC 2048
#define BN_EPS_C 1e-5f

__device__ __forceinline__ u16 f2bf(float f) {
  u32 b = __float_as_uint(f);
  b += 0x7FFFu + ((b >> 16) & 1u);          // round-to-nearest-even
  return (u16)(b >> 16);
}
__device__ __forceinline__ float bf2f(u16 h) {
  return __uint_as_float(((u32)h) << 16);
}

union FragBF { uint4 q[2]; v16bf v; };

// ---------------- embedding gather: h = bf16(emb[x]) ----------------
__global__ __launch_bounds__(256) void embed_kernel(const int* __restrict__ x,
    const float* __restrict__ emb, u16* __restrict__ h, int n) {
  size_t tid = (size_t)blockIdx.x * 256 + threadIdx.x;
  if (tid >= (size_t)n * HID) return;
  int row = (int)(tid >> 8), col = (int)(tid & 255);
  h[tid] = f2bf(emb[(size_t)x[row] * HID + col]);
}

// ---------------- edge scatter-add: agg[dst] += h[src] (fp32 atomics) ----------------
__global__ __launch_bounds__(256) void scatter_kernel(const u16* __restrict__ h,
    const int* __restrict__ src, const int* __restrict__ dst,
    float* __restrict__ agg, int E) {
  size_t tid = (size_t)blockIdx.x * 256 + threadIdx.x;
  int e = (int)(tid >> 6);
  if (e >= E) return;
  int j = (int)(tid & 63) << 2;
  int s = src[e], d = dst[e];
  uint2 hv = *(const uint2*)(h + (size_t)s * HID + j);   // 4 bf16
  float* ap = agg + (size_t)d * HID + j;
  atomicAdd(ap + 0, bf2f((u16)(hv.x & 0xFFFFu)));
  atomicAdd(ap + 1, bf2f((u16)(hv.x >> 16)));
  atomicAdd(ap + 2, bf2f((u16)(hv.y & 0xFFFFu)));
  atomicAdd(ap + 3, bf2f((u16)(hv.y >> 16)));
}

// ---------------- o0 = bf16(h + agg) ----------------
__global__ __launch_bounds__(256) void addcvt_kernel(const u16* __restrict__ h,
    const float* __restrict__ agg, u16* __restrict__ o, size_t total) {
  size_t tid = (size_t)blockIdx.x * 256 + threadIdx.x;
  if (tid >= total) return;
  o[tid] = f2bf(bf2f(h[tid]) + agg[tid]);
}

// ---------------- masked gather for the last layer ----------------
__global__ __launch_bounds__(256) void maskgather_kernel(const u16* __restrict__ h,
    const float* __restrict__ agg, const int* __restrict__ mask,
    u16* __restrict__ o0m, u16* __restrict__ hm, int M) {
  size_t tid = (size_t)blockIdx.x * 256 + threadIdx.x;
  if (tid >= (size_t)M * HID) return;
  int row = (int)(tid >> 8), col = (int)(tid & 255);
  size_t s = (size_t)mask[row] * HID + col;
  u16 hv = h[s];
  hm[tid]  = hv;
  o0m[tid] = f2bf(bf2f(hv) + agg[s]);
}

// ---------------- pack W (fp32 KxN row-major) into WMMA B-fragment bf16 layout ----------------
// out[((kb*N + n)*2 + half)*8 + v] packs W[kb*32+kk][n], W[..+1][n] where
// kk = (v<4 ? 2v : 16+2(v-4)) + 8*half  (ISA 16-bit 32x16 B layout)
__global__ __launch_bounds__(256) void packw_kernel(const float* __restrict__ W,
    u32* __restrict__ out, int K, int N) {
  size_t tid = (size_t)blockIdx.x * 256 + threadIdx.x;
  size_t total = (size_t)K * N / 2;
  if (tid >= total) return;
  int v    = (int)(tid & 7);
  int half = (int)((tid >> 3) & 1);
  size_t rem = tid >> 4;
  int n  = (int)(rem % N);
  int kb = (int)(rem / N);
  int kk = (v < 4 ? 2 * v : 16 + 2 * (v - 4)) + 8 * half;
  int k  = kb * 32 + kk;
  u32 lo = f2bf(W[(size_t)k * N + n]);
  u32 hi = f2bf(W[(size_t)(k + 1) * N + n]);
  out[tid] = lo | (hi << 16);
}

// ---------------- WMMA GEMM: C = A[Mx256] @ W[256xN] with fused epilogue ----------------
// MODE 0: bn+prelu -> bf16 ; MODE 1: bn+prelu + bf16 residual -> bf16
// MODE 2: bn+prelu -> f32  ; MODE 3: f32 out += A@W (no epilogue)
template<int MODE>
__global__ __launch_bounds__(256) void gemm_kernel(
    const u16* __restrict__ A, const u32* __restrict__ Bp,
    int M, int N,
    const float* __restrict__ bias, const float* __restrict__ g,
    const float* __restrict__ be,   const float* __restrict__ mean,
    const float* __restrict__ var,  const float* __restrict__ alpha,
    const u16* __restrict__ res, u16* __restrict__ outb, float* __restrict__ outf) {
  const int lane = threadIdx.x & 31;
  const int wave = threadIdx.x >> 5;
  const int r    = lane & 15;
  const int half = lane >> 4;
  const int rowBase = (blockIdx.x * 8 + wave) * 16;
  const int colBase = blockIdx.y * 64;

  int rowL = rowBase + r;
  if (rowL >= M) rowL = M - 1;                 // clamp (no divergence around WMMA)
  const u16* arow = A + (size_t)rowL * HID;

  v8f z = {0.f,0.f,0.f,0.f,0.f,0.f,0.f,0.f};
  v8f c[4]; c[0] = z; c[1] = z; c[2] = z; c[3] = z;

  for (int kb = 0; kb < 8; ++kb) {             // K = 256 = 8 * 32
    FragBF a;
    const u16* ap = arow + kb * 32 + half * 8; // ISA 16-bit 16x32 A layout
    a.q[0] = *(const uint4*)(ap);
    a.q[1] = *(const uint4*)(ap + 16);
#pragma unroll
    for (int t = 0; t < 4; ++t) {              // reuse A across 4 N-tiles
      int n = colBase + t * 16 + r;
      const u32* bp = Bp + (((size_t)kb * N + n) * 2 + half) * 8;
      FragBF b;
      b.q[0] = *(const uint4*)(bp);
      b.q[1] = *(const uint4*)(bp + 4);
      c[t] = __builtin_amdgcn_wmma_f32_16x16x32_bf16(
          false, a.v, false, b.v, (short)0, c[t], false, false);
    }
  }

  const float al = (MODE < 3) ? alpha[0] : 0.f;
#pragma unroll
  for (int t = 0; t < 4; ++t) {
    int col = colBase + t * 16 + r;
    float sc = 0.f, sb = 0.f, bi = 0.f;
    if (MODE < 3) {
      bi = bias[col];
      sc = g[col] * rsqrtf(var[col] + BN_EPS_C);
      sb = be[col] - mean[col] * sc;
    }
#pragma unroll
    for (int vv = 0; vv < 8; ++vv) {
      int row = rowBase + vv + 8 * half;       // ISA f32 C/D layout
      if (row >= M) continue;
      size_t oidx = (size_t)row * N + col;
      float val = c[t][vv];
      if (MODE < 3) {
        val = (val + bi) * sc + sb;            // batchnorm
        val = val > 0.f ? val : al * val;      // prelu
      }
      if      (MODE == 0) outb[oidx] = f2bf(val);
      else if (MODE == 1) outb[oidx] = f2bf(val + bf2f(res[oidx]));
      else if (MODE == 2) outf[oidx] = val;
      else                outf[oidx] += val;
    }
  }
}

// ---------------- log_softmax over rows of [M, 2048] ----------------
__global__ __launch_bounds__(256) void lsm_kernel(const float* __restrict__ logits,
    float* __restrict__ out) {
  __shared__ float sd[256];
  int row = blockIdx.x;
  const float* lr = logits + (size_t)row * VOC;
  float mx = -3.4e38f;
  for (int c = threadIdx.x; c < VOC; c += 256) mx = fmaxf(mx, lr[c]);
  sd[threadIdx.x] = mx; __syncthreads();
  for (int s = 128; s > 0; s >>= 1) {
    if (threadIdx.x < s) sd[threadIdx.x] = fmaxf(sd[threadIdx.x], sd[threadIdx.x + s]);
    __syncthreads();
  }
  mx = sd[0]; __syncthreads();
  float sum = 0.f;
  for (int c = threadIdx.x; c < VOC; c += 256) sum += expf(lr[c] - mx);
  sd[threadIdx.x] = sum; __syncthreads();
  for (int s = 128; s > 0; s >>= 1) {
    if (threadIdx.x < s) sd[threadIdx.x] += sd[threadIdx.x + s];
    __syncthreads();
  }
  float lse = mx + logf(sd[0]);
  for (int c = threadIdx.x; c < VOC; c += 256)
    out[(size_t)row * VOC + c] = lr[c] - lse;
}

extern "C" void kernel_launch(void* const* d_in, const int* in_sizes, int n_in,
                              void* d_out, int out_size, void* d_ws, size_t ws_size,
                              hipStream_t stream) {
  (void)n_in; (void)out_size; (void)ws_size;
  const int*   x    = (const int*)d_in[0];
  const int*   edge = (const int*)d_in[1];
  const int*   mask = (const int*)d_in[2];
  const float* emb  = (const float*)d_in[3];
  const int Nn = in_sizes[0];          // nodes (x is [N,1])
  const int E  = in_sizes[1] / 2;      // edges
  const int Mm = in_sizes[2];          // masked rows

  // JAX sorted-key flatten order within each layer:
  // W1, W2, [Wres], a1, a2, b1, b2, be1, be2, g1, g2, m1, m2, v1, v2
  enum { W1=0, W2=1, WRES=2, A1=3, A2=4, B1=5, B2=6, BE1=7, BE2=8,
         G1=9, G2=10, Mn1=11, Mn2=12, V1=13, V2=14 };
  auto LP = [&](int L, int which) -> const float* {
    int base = 4 + L * 14;
    bool wres = (L == 4);
    int idx;
    if (which <= W2)        idx = base + which;
    else if (which == WRES) idx = base + 2;
    else                    idx = base + which - (wres ? 0 : 1);
    return (const float*)d_in[idx];
  };

  // workspace carve (256B aligned)
  char* wsb = (char*)d_ws;
  size_t off = 0;
  auto carve = [&](size_t bytes) -> void* {
    void* p = wsb + off;
    off += (bytes + 255) & ~(size_t)255;
    return p;
  };
  u16*   h0     = (u16*)  carve((size_t)Nn * HID * 2);
  u16*   h1     = (u16*)  carve((size_t)Nn * HID * 2);
  float* agg    = (float*)carve((size_t)Nn * HID * 4);
  u16*   t0     = (u16*)  carve((size_t)Nn * HID * 2);
  u16*   t1     = (u16*)  carve((size_t)Nn * HID * 2);
  u16*   o0m    = (u16*)  carve((size_t)Mm * HID * 2);
  u16*   hm     = (u16*)  carve((size_t)Mm * HID * 2);
  u16*   t1m    = (u16*)  carve((size_t)Mm * HID * 2);
  float* logits = (float*)carve((size_t)Mm * VOC * 4);
  u32* w1p[5]; u32* w2p[5]; u32* wrp;
  for (int L = 0; L < 5; ++L) {
    w1p[L] = (u32*)carve((size_t)HID * HID * 2);
    int dout = (L == 4) ? VOC : HID;
    w2p[L] = (u32*)carve((size_t)HID * dout * 2);
  }
  wrp = (u32*)carve((size_t)HID * VOC * 2);

  // pack all weights into WMMA B layout (deterministic, every call)
  auto pack = [&](const float* W, u32* outp, int K, int N) {
    size_t total = (size_t)K * N / 2;
    packw_kernel<<<dim3((unsigned)((total + 255) / 256)), dim3(256), 0, stream>>>(W, outp, K, N);
  };
  for (int L = 0; L < 5; ++L) {
    pack(LP(L, W1), w1p[L], HID, HID);
    pack(LP(L, W2), w2p[L], HID, (L == 4) ? VOC : HID);
  }
  pack(LP(4, WRES), wrp, HID, VOC);

  { // embedding
    size_t total = (size_t)Nn * HID;
    embed_kernel<<<dim3((unsigned)((total + 255) / 256)), dim3(256), 0, stream>>>(x, emb, h0, Nn);
  }

  u16* h = h0; u16* hn = h1;
  dim3 blk(256);
  for (int L = 0; L < 5; ++L) {
    hipMemsetAsync(agg, 0, (size_t)Nn * HID * 4, stream);
    size_t sthreads = (size_t)E * 64;
    scatter_kernel<<<dim3((unsigned)((sthreads + 255) / 256)), blk, 0, stream>>>(
        h, edge, edge + E, agg, E);
    if (L < 4) {
      size_t total = (size_t)Nn * HID;
      addcvt_kernel<<<dim3((unsigned)((total + 255) / 256)), blk, 0, stream>>>(h, agg, t0, total);
      dim3 gg((Nn + 127) / 128, HID / 64);
      gemm_kernel<0><<<gg, blk, 0, stream>>>(t0, w1p[L], Nn, HID,
          LP(L,B1), LP(L,G1), LP(L,BE1), LP(L,Mn1), LP(L,V1), LP(L,A1),
          nullptr, t1, nullptr);
      gemm_kernel<1><<<gg, blk, 0, stream>>>(t1, w2p[L], Nn, HID,
          LP(L,B2), LP(L,G2), LP(L,BE2), LP(L,Mn2), LP(L,V2), LP(L,A2),
          h, hn, nullptr);
      u16* tmp = h; h = hn; hn = tmp;
    } else {
      // final layer only at masked rows (output consumed only there)
      size_t total = (size_t)Mm * HID;
      maskgather_kernel<<<dim3((unsigned)((total + 255) / 256)), blk, 0, stream>>>(
          h, agg, mask, o0m, hm, Mm);
      dim3 gg1((Mm + 127) / 128, HID / 64);
      gemm_kernel<0><<<gg1, blk, 0, stream>>>(o0m, w1p[L], Mm, HID,
          LP(L,B1), LP(L,G1), LP(L,BE1), LP(L,Mn1), LP(L,V1), LP(L,A1),
          nullptr, t1m, nullptr);
      dim3 gg2((Mm + 127) / 128, VOC / 64);
      gemm_kernel<2><<<gg2, blk, 0, stream>>>(t1m, w2p[L], Mm, VOC,
          LP(L,B2), LP(L,G2), LP(L,BE2), LP(L,Mn2), LP(L,V2), LP(L,A2),
          nullptr, nullptr, logits);
      gemm_kernel<3><<<gg2, blk, 0, stream>>>(hm, wrp, Mm, VOC,
          nullptr, nullptr, nullptr, nullptr, nullptr, nullptr,
          nullptr, nullptr, logits);
      lsm_kernel<<<dim3(Mm), blk, 0, stream>>>(logits, (float*)d_out);
    }
  }
}
